// DiscoBoxv2Head_42597485641750
// MI455X (gfx1250) — 8-bit matrix operand order repaired and checked by reading, compiled, not run on hardware
//
#include <hip/hip_runtime.h>
#include <hip/hip_bf16.h>
#include <stdint.h>

typedef __attribute__((ext_vector_type(16))) int   v16i;
typedef __attribute__((ext_vector_type(8)))  float v8f;

#define N_MASKS 256
#define HW      16384
#define NI      64
#define H       128
#define W       128

__global__ void pack_fp8_kernel(const float* __restrict__ m,
                                unsigned int* __restrict__ out, int n4) {
    int i = blockIdx.x * blockDim.x + threadIdx.x;
    if (i >= n4) return;
    const float4 v = reinterpret_cast<const float4*>(m)[i];
    unsigned int b = (v.x > 0.5f ? 0x38u : 0u)
                   | ((v.y > 0.5f ? 0x38u : 0u) << 8)
                   | ((v.z > 0.5f ? 0x38u : 0u) << 16)
                   | ((v.w > 0.5f ? 0x38u : 0u) << 24);
    out[i] = b;
}

__device__ __forceinline__ v16i load_a_chunk(const unsigned char* p) {
    v16i va;
#pragma unroll
    for (int j = 0; j < 8; ++j) {
        uint2 d = *reinterpret_cast<const uint2*>(p + 16 * j);
        va[2 * j]     = (int)d.x;
        va[2 * j + 1] = (int)d.y;
    }
    return va;
}

__device__ __forceinline__ v16i load_b_chunk(const unsigned char* p) {
    v16i vb;
#pragma unroll
    for (int g = 0; g < 8; ++g) {
        uint2 d = *reinterpret_cast<const uint2*>(p + 16 * g);
        vb[2 * g]     = (int)d.x;
        vb[2 * g + 1] = (int)d.y;
    }
    return vb;
}

__global__ __launch_bounds__(128)
void gram_wmma_kernel(const unsigned char* __restrict__ A,
                      float* __restrict__ inter) {
    const int lane = threadIdx.x & 31;
    const int wave = threadIdx.x >> 5;
    const int tile = blockIdx.x;
    const int iT = tile >> 4;
    const int jT = tile & 15;

    const int rowA  = iT * 16 + (lane & 15);
    const int rowB  = jT * 16 + (lane & 15);
    const int halfA = (lane & 16) ? 8 : 0;
    const int halfB = (lane & 16) ? 8 : 0;
    const int k0    = wave * 4096;
    const unsigned char* pA = A + (size_t)rowA * HW + halfA + k0;
    const unsigned char* pB = A + (size_t)rowB * HW + halfB + k0;

    v8f acc = {};
    v16i va0 = load_a_chunk(pA);
    v16i vb0 = load_b_chunk(pB);
    v16i va1, vb1;

    for (int kk = 0; kk < 4096; kk += 256) {
        va1 = load_a_chunk(pA + kk + 128);
        vb1 = load_b_chunk(pB + kk + 128);
        acc = __builtin_amdgcn_wmma_f32_16x16x128_fp8_fp8(va0, vb0, (short)0, acc,
                                                          false, false);
        if (kk + 256 < 4096) {
            va0 = load_a_chunk(pA + kk + 256);
            vb0 = load_b_chunk(pB + kk + 256);
        }
        acc = __builtin_amdgcn_wmma_f32_16x16x128_fp8_fp8(va1, vb1, (short)0, acc,
                                                          false, false);
    }

    __shared__ float part[4][32][8];
#pragma unroll
    for (int v = 0; v < 8; ++v) part[wave][lane][v] = acc[v];
    __syncthreads();
    if (wave == 0) {
        const int n  = lane & 15;
        const int mb = (lane & 16) ? 8 : 0;
#pragma unroll
        for (int v = 0; v < 8; ++v) {
            float s = part[0][lane][v] + part[1][lane][v]
                    + part[2][lane][v] + part[3][lane][v];
            inter[(size_t)(iT * 16 + mb + v) * N_MASKS + jT * 16 + n] = s;
        }
    }
}

__device__ __forceinline__ float decay_iou_ij(const float* inter,
                                              const int* labels, int i, int j) {
    if (i < j && labels[i] == labels[j]) {
        float si = inter[i * N_MASKS + i];
        float sj = inter[j * N_MASKS + j];
        float in_ = inter[i * N_MASKS + j];
        return in_ / (si + sj - in_);
    }
    return 0.0f;
}

__global__ __launch_bounds__(256)
void nms_comp_kernel(const float* __restrict__ inter,
                     const int* __restrict__ labels, float* __restrict__ comp) {
    const int j = blockIdx.x, i = threadIdx.x;
    float d = decay_iou_ij(inter, labels, i, j);
    __shared__ float red[256];
    red[i] = d; __syncthreads();
    for (int s = 128; s > 0; s >>= 1) {
        if (i < s) red[i] = fmaxf(red[i], red[i + s]);
        __syncthreads();
    }
    if (i == 0) comp[j] = red[0];
}

__global__ __launch_bounds__(256)
void nms_coef_kernel(const float* __restrict__ inter,
                     const int* __restrict__ labels,
                     const float* __restrict__ comp,
                     const float* __restrict__ cate_scores,
                     float* __restrict__ scores_out) {
    const int j = blockIdx.x, i = threadIdx.x;
    float d  = decay_iou_ij(inter, labels, i, j);
    float ci = comp[i];
    float v  = __expf(2.0f * (ci * ci - d * d));
    __shared__ float red[256];
    red[i] = v; __syncthreads();
    for (int s = 128; s > 0; s >>= 1) {
        if (i < s) red[i] = fminf(red[i], red[i + s]);
        __syncthreads();
    }
    if (i == 0) scores_out[j] = cate_scores[j] * red[0];
}

__global__ __launch_bounds__(256)
void crf_pre_kernel(const float* __restrict__ feat, float* __restrict__ kern) {
    const int b = blockIdx.x >> 6;
    const int p = ((blockIdx.x & 63) << 8) + threadIdx.x;
    const int cy = p >> 7, cx = p & 127;
    const float* fb = feat + (size_t)b * 3 * HW;
    const float f0 = fb[p] + 10.f, f1 = fb[HW + p] + 10.f, f2 = fb[2 * HW + p] + 10.f;
    float* kb = kern + (size_t)b * 9 * HW;
#pragma unroll
    for (int t = 0; t < 9; ++t) {
        const int dy = t / 3 - 1, dx = t % 3 - 1;
        const int ny = cy + dy, nx = cx + dx;
        float n0 = 0.f, n1 = 0.f, n2 = 0.f;
        if (ny >= 0 && ny < H && nx >= 0 && nx < W) {
            const int q = (ny << 7) + nx;
            n0 = fb[q] + 10.f; n1 = fb[HW + q] + 10.f; n2 = fb[2 * HW + q] + 10.f;
        }
        const float ds = (n0 - f0) * (n0 - f0) + (n1 - f1) * (n1 - f1)
                       + (n2 - f2) * (n2 - f2);
        const float sp = (float)(dy * dy + dx * dx);
        kb[t * HW + p] = 3.0f * __expf(-2.0f * ds - sp * (1.0f / 1800.0f));
    }
}

__global__ __launch_bounds__(1024)
void crf_iter_kernel(const float* __restrict__ x, const float* __restrict__ tgt,
                     const float* __restrict__ kern,
                     float* __restrict__ masks_out, float* __restrict__ valid_out) {
    const int b = blockIdx.x;
    __shared__ unsigned char st[HW];
    __shared__ float red[1024];
    const float* xb = x   + (size_t)b * HW;
    const float* tb = tgt + (size_t)b * HW;
    const float* kb = kern + (size_t)b * 9 * HW;

    for (int p = threadIdx.x; p < HW; p += 1024) {
        const float xv = xb[p] * tb[p];
        st[p] = (unsigned char)((xv > 0.5f ? 1 : 0) | (tb[p] > 0.5f ? 2 : 0));
    }
    __syncthreads();

    const float L0 = 0.79850769f;
    const float L1 = 0.59783700f;

    for (int it = 0; it < 10; ++it) {
        unsigned char ns[16];
#pragma unroll 1
        for (int k = 0; k < 16; ++k) {
            const int p = threadIdx.x + (k << 10);
            const int cy = p >> 7, cx = p & 127;
            float a0 = 0.f, a1 = 0.f;
#pragma unroll
            for (int t = 0; t < 9; ++t) {
                const int dy = t / 3 - 1, dx = t % 3 - 1;
                const int ny = cy + dy, nx = cx + dx;
                if (ny >= 0 && ny < H && nx >= 0 && nx < W) {
                    const unsigned char sv = st[(ny << 7) + nx];
                    const float kv = kb[t * HW + p];
                    a1 += kv * ((sv & 1) ? L1 : L0);
                    a0 += kv * ((sv & 1) ? L0 : L1);
                }
            }
            const float f0 = __expf(-a0) + 1e-6f;
            const float tg = (st[p] & 2) ? 1.f : 0.f;
            const float f1 = __expf(-a1) * tg + 1e-6f;
            ns[k] = (f1 > f0) ? 1 : 0;
        }
        __syncthreads();
#pragma unroll
        for (int k = 0; k < 16; ++k) {
            const int p = threadIdx.x + (k << 10);
            st[p] = (unsigned char)((st[p] & 2) | ns[k]);
        }
        __syncthreads();
    }

    int cnt = 0;
#pragma unroll
    for (int k = 0; k < 16; ++k) {
        const int p = threadIdx.x + (k << 10);
        const int s = st[p] & 1;
        masks_out[(size_t)b * HW + p] = (float)s;
        cnt += s;
    }
    red[threadIdx.x] = (float)cnt; __syncthreads();
    for (int s = 512; s > 0; s >>= 1) {
        if (threadIdx.x < s) red[threadIdx.x] += red[threadIdx.x + s];
        __syncthreads();
    }
    if (threadIdx.x == 0) {
        const float c = red[0];
        valid_out[b] = (c >= HW * 0.05f && c <= HW * 0.95f) ? 1.f : 0.f;
    }
}

extern "C" void kernel_launch(void* const* d_in, const int* in_sizes, int n_in,
                              void* d_out, int out_size, void* d_ws, size_t ws_size,
                              hipStream_t stream) {
    const float* seg         = (const float*)d_in[0];
    const float* cate_scores = (const float*)d_in[1];
    const float* feat        = (const float*)d_in[2];
    const float* x           = (const float*)d_in[3];
    const float* tgt         = (const float*)d_in[4];
    const int*   labels      = (const int*)d_in[5];

    float* out_scores = (float*)d_out;
    float* out_masks  = out_scores + N_MASKS;
    float* out_valid  = out_masks + (size_t)NI * HW;

    unsigned char* packA = (unsigned char*)d_ws;
    float* inter = (float*)(packA + (size_t)N_MASKS * HW);
    float* comp  = inter + N_MASKS * N_MASKS;
    float* kern  = comp + N_MASKS;

    {
        const int n4 = (N_MASKS * HW) / 4;
        pack_fp8_kernel<<<(n4 + 255) / 256, 256, 0, stream>>>(
            seg, (unsigned int*)packA, n4);
    }
    gram_wmma_kernel<<<N_MASKS, 128, 0, stream>>>(packA, inter);
    nms_comp_kernel<<<N_MASKS, 256, 0, stream>>>(inter, labels, comp);
    nms_coef_kernel<<<N_MASKS, 256, 0, stream>>>(inter, labels, comp,
                                                 cate_scores, out_scores);
    crf_pre_kernel<<<NI * 64, 256, 0, stream>>>(feat, kern);
    crf_iter_kernel<<<NI, 1024, 0, stream>>>(x, tgt, kern, out_masks, out_valid);
}
